// AttentionAutoInt_12953621364957
// MI455X (gfx1250) — matrix operand order repaired and checked by reading, not compiled, not run on hardware
//
#include <hip/hip_runtime.h>
#include <math.h>

typedef __attribute__((ext_vector_type(16))) _Float16 v16h;
typedef __attribute__((ext_vector_type(8)))  _Float16 v8h;
typedef __attribute__((ext_vector_type(4)))  _Float16 v4h;
typedef __attribute__((ext_vector_type(8)))  float    v8f;

#define NTOK 64     // M: fields per batch
#define DIM  128    // D = DP

// LDS byte offsets
#define OFF_XH 0        // f16 [64][128]  x (f16)
#define OFF_QH 16384    // f16 [64][128]  q
#define OFF_KH 32768    // f16 [64][128]  k
#define OFF_VT 49152    // f16 [128][64]  v transposed (e-major)
#define OFF_SC 65536    // f32 [64][64]   scores / exp
#define OFF_PR 81920    // f16 [64][64]   softmax probs
#define SMEM_BYTES 90112

// ---- WMMA fragment loaders (wave32, 16x16x32 f16) -------------------------
// A-matrix 16x32: lane holds row m=lane&15; elements 0..15 -> K=h*16..h*16+15
// (h = lane>>4).  rowp = base + m*ld + kbase.
__device__ __forceinline__ v16h load_a_frag(const _Float16* rowp, int h) {
    return *(const v16h*)(rowp + h * 16);
}
// B-matrix 32x16: lane holds col n=lane&15; 16 consecutive K values,
// lanes 0-15 -> K=0..15, lanes 16-31 -> K=16..31.  colp = base + n*ld + kbase,
// storage must be K-contiguous per column.
__device__ __forceinline__ v16h load_b_frag(const _Float16* colp, int h) {
    return *(const v16h*)(colp + h * 16);
}

__device__ __forceinline__ v8f wmma_f16(v16h a, v16h b, v8f c) {
    return __builtin_amdgcn_wmma_f32_16x16x32_f16(
        /*neg_a=*/false, a, /*neg_b=*/false, b,
        /*c_mod=*/(short)0, c, /*reuse_a=*/false, /*reuse_b=*/false);
}

// ---- Pre-kernel: f32 weights -> f16, transposed (e-major) -----------------
// wt[w][e][d] = (f16) W_w[d][e] ; 4 * 128 * 128 = 65536 elements
__global__ void autoint_cvt_weights(const float* __restrict__ wq,
                                    const float* __restrict__ wk,
                                    const float* __restrict__ wv,
                                    const float* __restrict__ wr,
                                    _Float16* __restrict__ wt) {
    int g = blockIdx.x * blockDim.x + threadIdx.x;     // 0..65535
    int w = g >> 14;
    int rem = g & 16383;
    int e = rem >> 7;
    int d = rem & 127;
    const float* src = (w == 0) ? wq : (w == 1) ? wk : (w == 2) ? wv : wr;
    wt[(size_t)w * (DIM * DIM) + e * DIM + d] = (_Float16)src[d * DIM + e];
}

// ---- Main fused attention kernel: 1 block = 1 batch, 256 threads = 8 waves
__global__ void autoint_attn(const float* __restrict__ x,
                             const _Float16* __restrict__ wt,   // [4][128][128] e-major f16
                             float* __restrict__ out) {
    extern __shared__ char smem[];
    _Float16* XH = (_Float16*)(smem + OFF_XH);
    _Float16* QH = (_Float16*)(smem + OFF_QH);
    _Float16* KH = (_Float16*)(smem + OFF_KH);
    _Float16* VT = (_Float16*)(smem + OFF_VT);
    float*    SC = (float*)   (smem + OFF_SC);
    _Float16* PR = (_Float16*)(smem + OFF_PR);

    const int tid  = threadIdx.x;
    const int lane = tid & 31;
    const int wid  = tid >> 5;
    const int h    = lane >> 4;
    const int r16  = lane & 15;
    const int b    = blockIdx.x;
    const int n    = wid * 16 + r16;   // output column owned by this lane in
                                       // stages A and D (nt == wid mapping)

    // r projection tiles (mt = 0..3, nt = wid) stay resident in VGPRs:
    // same wave/lane/element produces them in stage A and consumes in stage D.
    v8f racc[4];

    // ---- load x[b] (64x128 f32) -> XH (f16) -------------------------------
    {
        const float4* xg = (const float4*)(x + (size_t)b * (NTOK * DIM));
#pragma unroll
        for (int t = 0; t < 8; ++t) {
            int idx = tid + t * 256;                   // 2048 float4 total
            float4 f = xg[idx];
            v4h hv;
            hv.x = (_Float16)f.x; hv.y = (_Float16)f.y;
            hv.z = (_Float16)f.z; hv.w = (_Float16)f.w;
            *(v4h*)(XH + idx * 4) = hv;
        }
    }
    __syncthreads();

    // ---- Stage A: q,k,v,r projections (x @ W_w) ---------------------------
    // Mapping: mt = j, nt = wid for every matrix.
    // q/k: one tile in flight at a time (unroll 1) to keep VGPR pressure low.
#pragma unroll 1
    for (int w = 0; w < 2; ++w) {
        const _Float16* wbase = wt + (size_t)w * (DIM * DIM);
        _Float16* dst = (w == 0) ? QH : KH;
#pragma unroll 1
        for (int j = 0; j < 4; ++j) {
            v8f acc = {};
#pragma unroll
            for (int kk = 0; kk < 4; ++kk) {
                int kb = kk * 32;
                v16h a  = load_a_frag(XH + (j * 16 + r16) * DIM + kb, h);
                v16h bb = load_b_frag(wbase + n * DIM + kb, h);
                acc = wmma_f16(a, bb, acc);
            }
#pragma unroll
            for (int i = 0; i < 8; ++i)
                dst[(j * 16 + i + 8 * h) * DIM + n] = (_Float16)acc[i];
        }
    }
    // v: stored transposed; lane's 8 elements are m = j*16+8h..+7, contiguous
    // in VT[n][m] -> one 16-byte DS store per tile.
    {
        const _Float16* wbase = wt + (size_t)2 * (DIM * DIM);
#pragma unroll 1
        for (int j = 0; j < 4; ++j) {
            v8f acc = {};
#pragma unroll
            for (int kk = 0; kk < 4; ++kk) {
                int kb = kk * 32;
                v16h a  = load_a_frag(XH + (j * 16 + r16) * DIM + kb, h);
                v16h bb = load_b_frag(wbase + n * DIM + kb, h);
                acc = wmma_f16(a, bb, acc);
            }
            union { v8h v; _Float16 e[8]; } pk;
#pragma unroll
            for (int i = 0; i < 8; ++i) pk.e[i] = (_Float16)acc[i];
            *(v8h*)(VT + n * NTOK + j * 16 + 8 * h) = pk.v;
        }
    }
    // r: unrolled so racc[] indices are compile-time; tiles stay in VGPRs.
    {
        const _Float16* wbase = wt + (size_t)3 * (DIM * DIM);
#pragma unroll
        for (int j = 0; j < 4; ++j) {
            v8f acc = {};
#pragma unroll
            for (int kk = 0; kk < 4; ++kk) {
                int kb = kk * 32;
                v16h a  = load_a_frag(XH + (j * 16 + r16) * DIM + kb, h);
                v16h bb = load_b_frag(wbase + n * DIM + kb, h);
                acc = wmma_f16(a, bb, acc);
            }
            racc[j] = acc;
        }
    }
    __syncthreads();

    // ---- Stage B: scores = q @ k^T  (B-fragment columns are rows of k) ----
#pragma unroll 1
    for (int tt = 0; tt < 2; ++tt) {
        int t  = wid * 2 + tt;
        int mt = t >> 2;
        int nt = t & 3;
        v8f acc = {};
#pragma unroll
        for (int kk = 0; kk < 4; ++kk) {
            int kb = kk * 32;
            v16h a  = load_a_frag(QH + (mt * 16 + r16) * DIM + kb, h);
            v16h bb = load_b_frag(KH + (nt * 16 + r16) * DIM + kb, h);
            acc = wmma_f16(a, bb, acc);
        }
        int sn = nt * 16 + r16;
#pragma unroll
        for (int i = 0; i < 8; ++i) {
            int m = mt * 16 + i + 8 * h;
            SC[m * NTOK + sn] = acc[i];
        }
    }
    __syncthreads();

    // ---- Softmax over rows of SC; probs -> PR (f16) -----------------------
    {
        int row = tid >> 2;                // 64 rows, 4 threads/row (same wave)
        int qo  = (tid & 3) * 16;
        float* srow = SC + row * NTOK;
        float mx = srow[0];
        for (int j = 1; j < NTOK; ++j) mx = fmaxf(mx, srow[j]);
        __syncthreads();
#pragma unroll
        for (int j = 0; j < 16; ++j)
            srow[qo + j] = __expf(srow[qo + j] - mx);
        __syncthreads();
        float sum = 0.f;
        for (int j = 0; j < NTOK; ++j) sum += srow[j];
        float inv = 1.0f / sum;
#pragma unroll
        for (int j = 0; j < 16; ++j)
            PR[row * NTOK + qo + j] = (_Float16)(srow[qo + j] * inv);
    }
    __syncthreads();

    // ---- Stage D: out = relu(r + probs @ v), out tile (mt=tt, nt=wid) -----
    // B-fragments (VT + n*NTOK) are identical for all tt -> hoisted once.
    float* og = out + (size_t)b * (NTOK * DIM);
#pragma unroll
    for (int tt = 0; tt < 4; ++tt) {
        v8f acc = {};
#pragma unroll
        for (int kk = 0; kk < 2; ++kk) {
            int kb = kk * 32;
            v16h a  = load_a_frag(PR + (tt * 16 + r16) * NTOK + kb, h);
            v16h bb = load_b_frag(VT + n * NTOK + kb, h);
            acc = wmma_f16(a, bb, acc);
        }
#pragma unroll
        for (int i = 0; i < 8; ++i) {
            int m = tt * 16 + i + 8 * h;
            og[m * DIM + n] = fmaxf(racc[tt][i] + acc[i], 0.0f);
        }
    }
}

extern "C" void kernel_launch(void* const* d_in, const int* in_sizes, int n_in,
                              void* d_out, int out_size, void* d_ws, size_t ws_size,
                              hipStream_t stream) {
    (void)in_sizes; (void)n_in; (void)out_size; (void)ws_size;
    const float* x  = (const float*)d_in[0];
    const float* wq = (const float*)d_in[1];
    const float* wk = (const float*)d_in[2];
    const float* wv = (const float*)d_in[3];
    const float* wr = (const float*)d_in[4];
    _Float16* wt = (_Float16*)d_ws;            // 4*128*128 f16 = 128 KB
    float* out = (float*)d_out;

    autoint_cvt_weights<<<256, 256, 0, stream>>>(wq, wk, wv, wr, wt);
    autoint_attn<<<8192, 256, SMEM_BYTES, stream>>>(x, wt, out);
}
